// MinimalAttnHead_55929064128765
// MI455X (gfx1250) — matrix operand order repaired and checked by reading, not compiled, not run on hardware
//
#include <hip/hip_runtime.h>
#include <math.h>

// ---------- types for gfx1250 WMMA ----------
typedef __bf16 bf16_t;
typedef __attribute__((ext_vector_type(16))) __bf16 v16bf;
typedef __attribute__((ext_vector_type(8)))  __bf16 v8bf;
typedef __attribute__((ext_vector_type(8)))  float  f32x8;
typedef __attribute__((ext_vector_type(16))) float  f32x16;
typedef __attribute__((ext_vector_type(8)))  float  v8f;
typedef __attribute__((ext_vector_type(4)))  unsigned uint32x4;
typedef __attribute__((ext_vector_type(8)))  int      int32x8;
typedef __attribute__((ext_vector_type(4)))  int      int32x4;

#define WMMA_BF16(A_, B_, C_) \
  __builtin_amdgcn_wmma_f32_16x16x32_bf16(false, (A_), false, (B_), (short)0, (C_), false, false)

// Tensor Data Mover availability (this toolchain: 6-arg builtin,
// (uint32x4 g0, int32x8 g1, int32x4 g2, int32x4 g3, int32x8 g4, i32 cpol))
#if defined(__has_builtin)
#  if __has_builtin(__builtin_amdgcn_tensor_load_to_lds) && \
      __has_builtin(__builtin_amdgcn_s_wait_tensorcnt)
#    define HAVE_TDM 1
#  endif
#endif
#ifndef HAVE_TDM
#  define HAVE_TDM 0
#endif

static constexpr int NB   = 4;      // batch
static constexpr int NS   = 4096;   // sequence
static constexpr int ND   = 1024;   // model dim
static constexpr int NDQK = 64;     // head dim

// ===================================================================
// Kernel 1: convert encodings (fp32) -> V (bf16). 8 elems / thread.
// ===================================================================
__global__ __launch_bounds__(256) void convert_v_kernel(const float* __restrict__ enc,
                                                        bf16_t* __restrict__ Vb) {
  size_t i = ((size_t)blockIdx.x * 256 + threadIdx.x) * 8;
  f32x8 v = *(const f32x8*)(enc + i);
  *(v8bf*)(Vb + i) = __builtin_convertvector(v, v8bf);
}

// ===================================================================
// Kernel 2: Q = enc*Wq^T * 0.125 (bf16), K = enc*Wk^T (bf16)
// block = 128 threads (4 waves); wave w owns output cols [16w,16w+16)
// grid.x = (B*S)/16 row tiles, grid.y = {0:Q, 1:K}
// ===================================================================
__global__ __launch_bounds__(128) void proj_kernel(const float* __restrict__ enc,
                                                   const float* __restrict__ Wq,
                                                   const float* __restrict__ Wk,
                                                   bf16_t* __restrict__ Qb,
                                                   bf16_t* __restrict__ Kb) {
  const int wave = threadIdx.x >> 5;
  const int lane = threadIdx.x & 31;
  const int hlf  = lane >> 4;
  const int l15  = lane & 15;
  const int rowTile = blockIdx.x;
  const int proj    = blockIdx.y;       // 0 -> Q, 1 -> K
  const float* W = proj ? Wk : Wq;
  bf16_t* Out    = proj ? Kb : Qb;
  const int n0 = wave * 16;

  const float* encRow = enc + (size_t)(rowTile * 16 + l15) * ND; // A: row m = l&15
  const float* wRow   = W   + (size_t)(n0 + l15) * ND;           // B: col n = l&15

  v8f acc = {};
  for (int kb = 0; kb < ND; kb += 32) {
    // A (16x32): lane holds A[m][hlf*16 + 0..15]
    f32x8 a0 = *(const f32x8*)(encRow + kb + hlf * 16);
    f32x8 a1 = *(const f32x8*)(encRow + kb + hlf * 16 + 8);
    v8bf a0b = __builtin_convertvector(a0, v8bf);
    v8bf a1b = __builtin_convertvector(a1, v8bf);
    v16bf A = __builtin_shufflevector(a0b, a1b, 0,1,2,3,4,5,6,7,8,9,10,11,12,13,14,15);
    // B (32x16): lane holds B[hlf*16 + 0..15][n] = W[n][kb + hlf*16 + 0..15]
    f32x16 bw = *(const f32x16*)(wRow + kb + hlf * 16);
    v16bf Bv = __builtin_convertvector(bw, v16bf);
    acc = WMMA_BF16(A, Bv, acc);
  }
  const float scale = proj ? 1.0f : 0.125f;   // 1/sqrt(64) folded into Q
#pragma unroll
  for (int r = 0; r < 8; ++r) {
    int m = rowTile * 16 + r + hlf * 8;       // C layout: row = r + 8*(lane>>4)
    Out[(size_t)m * NDQK + n0 + l15] = (bf16_t)(acc[r] * scale);
  }
}

// ===================================================================
// Kernel 3: causal flash attention.
// block = 256 threads (8 waves): 32 queries x 256 output columns.
// waves 0-3: score tiles; wave 0: online softmax + TDM V staging;
// all 8 waves: P@V.   grid = (S/32, B, D/256)
// ===================================================================
#define BQ 32
#define BK 32
#define DN 256
#define SSTR 33   // padded f32 score row stride
#define PSTR 40   // padded bf16 P row stride (16B-aligned rows)
#define VPADE 8   // V row pad elements (16B = 4 DWORDs)
#define VSTR (DN + VPADE)  // 264

__global__ __launch_bounds__(256) void flash_kernel(const bf16_t* __restrict__ Qb,
                                                    const bf16_t* __restrict__ Kb,
                                                    const bf16_t* __restrict__ Vb,
                                                    float* __restrict__ Out) {
  __shared__ alignas(32) float  sS[BQ][SSTR];
  __shared__ alignas(32) bf16_t sP[BQ][PSTR];
  __shared__ alignas(32) bf16_t sV[BK][VSTR];
  __shared__ alignas(32) float  sAlpha[BQ];
  __shared__ alignas(32) float  sL[BQ];

  const int tid  = threadIdx.x;
  const int wave = tid >> 5;
  const int lane = tid & 31;
  const int hlf  = lane >> 4;
  const int l15  = lane & 15;

  const int b  = blockIdx.y;
  const int q0 = blockIdx.x * BQ;
  const int d0 = blockIdx.z * DN;

  const bf16_t* Qbase = Qb + (size_t)b * NS * NDQK;
  const bf16_t* Kbase = Kb + (size_t)b * NS * NDQK;
  const bf16_t* Vbase = Vb + (size_t)b * NS * ND;

  // Score-duty assignment for waves 0..3: query row-tile qr, key group kg.
  const int qr = wave >> 1;
  const int kg = wave & 1;
  v16bf Qa0 = {}, Qa1 = {};
  if (wave < 4) {
    const bf16_t* qrow = Qbase + (size_t)(q0 + qr * 16 + l15) * NDQK;
    v8bf q00 = *(const v8bf*)(qrow + hlf * 16);
    v8bf q01 = *(const v8bf*)(qrow + hlf * 16 + 8);
    Qa0 = __builtin_shufflevector(q00, q01, 0,1,2,3,4,5,6,7,8,9,10,11,12,13,14,15);
    v8bf q10 = *(const v8bf*)(qrow + 32 + hlf * 16);
    v8bf q11 = *(const v8bf*)(qrow + 40 + hlf * 16);
    Qa1 = __builtin_shufflevector(q10, q11, 0,1,2,3,4,5,6,7,8,9,10,11,12,13,14,15);
  }

#if HAVE_TDM
  // LDS byte offset of sV (low 32 bits of the generic pointer are the LDS offset)
  const unsigned ldsOffV = (unsigned)(uintptr_t)(&sV[0][0]);
#endif

  v8f acc[2][2] = {};                 // [query row-tile][col tile]; cols = (wave*2+c)*16
  float mRow = -3.0e38f, lRow = 0.0f; // wave-0 per-lane (lane == query row) softmax state

  const int qHi = q0 + BQ - 1;
  for (int kb = 0; kb <= qHi; kb += BK) {
    // ---- Phase A: score tiles (waves 0-3), scaled Q already ----
    if (wave < 4) {
      const int k0 = kb + kg * 16;
      const bf16_t* krow = Kbase + (size_t)(k0 + l15) * NDQK; // B col n = key k0+l15
      v16bf Kt0 = *(const v16bf*)(krow + hlf * 16);           // feats 0..31
      v16bf Kt1 = *(const v16bf*)(krow + 32 + hlf * 16);      // feats 32..63
      v8f s = {};
      s = WMMA_BF16(Qa0, Kt0, s);
      s = WMMA_BF16(Qa1, Kt1, s);
#pragma unroll
      for (int r = 0; r < 8; ++r) {
        int qg = q0 + qr * 16 + r + hlf * 8;
        int kgidx = k0 + l15;
        float v = (kgidx > qg) ? -1.0e9f : s[r];              // causal mask
        sS[qr * 16 + r + hlf * 8][kg * 16 + l15] = v;
      }
    }
#if HAVE_TDM
    // ---- Phase A': TDM DMA of V block [32 keys x 256 cols] -> padded LDS rows ----
    if (wave == 0) {
      unsigned long long ga = (unsigned long long)(uintptr_t)(Vbase + (size_t)kb * ND + d0);
      // D# group 0: count=1 | lds_addr | global_addr[56:0] | type=2
      uint32x4 g0;
      g0[0] = 1u;
      g0[1] = ldsOffV;
      g0[2] = (unsigned)(ga & 0xFFFFFFFFu);
      g0[3] = (unsigned)((ga >> 32) & 0x01FFFFFFu) | (2u << 30);
      // D# group 1: data_size=1 (2B), pad_enable, pad_interval=6 (128 DW = 512B rows),
      // pad_amount=3 (4 DW = 16B); tensor_dim0=1024, tensor_dim1=4096;
      // tile_dim0=256, tile_dim1=32; tensor_dim0_stride=1024.
      int32x8 g1;
      g1[0] = (1 << 16) | (1 << 20) | (6 << 22) | (3 << 25);
      g1[1] = (int)(1024u << 16);      // tensor_dim0[15:0] in bits 63:48
      g1[2] = (int)(4096u << 16);      // tensor_dim0[31:16]=0, tensor_dim1[15:0] in 95:80
      g1[3] = (int)(256u << 16);       // tensor_dim1[31:16]=0, tile_dim0 in 127:112
      g1[4] = 32;                      // tile_dim1=32, tile_dim2=0
      g1[5] = 1024;                    // tensor_dim0_stride[31:0]
      g1[6] = 0;                       // stride0[47:32]=0, dim1_stride low=0
      g1[7] = 0;
      int32x4 gz4 = {0, 0, 0, 0};
      int32x8 gz8 = {0, 0, 0, 0, 0, 0, 0, 0};
      __builtin_amdgcn_tensor_load_to_lds(g0, g1, gz4, gz4, gz8, 0);
      __builtin_amdgcn_s_wait_tensorcnt((short)0);
    }
#else
    // ---- Phase A' (fallback): stage V block via VGPRs (all threads) ----
    {
      int vrow = tid >> 3;
      int vcol = (tid & 7) * 32;
      const bf16_t* src = Vbase + (size_t)(kb + vrow) * ND + d0 + vcol;
      *(v8bf*)(&sV[vrow][vcol])      = *(const v8bf*)(src);
      *(v8bf*)(&sV[vrow][vcol + 8])  = *(const v8bf*)(src + 8);
      *(v8bf*)(&sV[vrow][vcol + 16]) = *(const v8bf*)(src + 16);
      *(v8bf*)(&sV[vrow][vcol + 24]) = *(const v8bf*)(src + 24);
    }
#endif
    __syncthreads();

    // ---- Phase B: online softmax, wave 0, lane = query row ----
    if (tid < 32) {
      const int r = tid;
      float mNew = mRow;
#pragma unroll
      for (int j = 0; j < BK; ++j) mNew = fmaxf(mNew, sS[r][j]);
      float alpha = __expf(mRow - mNew);
      float lNew = lRow * alpha;
#pragma unroll
      for (int j = 0; j < BK; ++j) {
        float p = __expf(sS[r][j] - mNew);
        lNew += p;
        sP[r][j] = (bf16_t)p;
      }
      mRow = mNew; lRow = lNew;
      sAlpha[r] = alpha;
      sL[r]     = lNew;
    }
    __syncthreads();

    // ---- Phase C: O = O*alpha + P @ V (all 8 waves) ----
    v16bf Vt[2];
#pragma unroll
    for (int c = 0; c < 2; ++c) {
      const int vn = (wave * 2 + c) * 16 + l15;   // B col n
      v16bf t = {};
#pragma unroll
      for (int j = 0; j < 16; ++j) t[j] = sV[hlf * 16 + j][vn];
      Vt[c] = t;
    }
#pragma unroll
    for (int t = 0; t < 2; ++t) {
      const bf16_t* prow = &sP[t * 16 + l15][0];
      v8bf p0 = *(const v8bf*)(prow + hlf * 16);
      v8bf p1 = *(const v8bf*)(prow + hlf * 16 + 8);
      v16bf Pa = __builtin_shufflevector(p0, p1, 0,1,2,3,4,5,6,7,8,9,10,11,12,13,14,15);
      f32x8 al = *(const f32x8*)(&sAlpha[t * 16 + hlf * 8]);
#pragma unroll
      for (int c = 0; c < 2; ++c) {
#pragma unroll
        for (int r = 0; r < 8; ++r) acc[t][c][r] *= al[r];
        acc[t][c] = WMMA_BF16(Pa, Vt[c], acc[t][c]);
      }
    }
    __syncthreads();
  }

  // ---- Epilogue: divide by softmax denominator, store fp32 ----
#pragma unroll
  for (int t = 0; t < 2; ++t) {
    f32x8 lr = *(const f32x8*)(&sL[t * 16 + hlf * 8]);
#pragma unroll
    for (int c = 0; c < 2; ++c) {
      int col = d0 + (wave * 2 + c) * 16 + l15;
#pragma unroll
      for (int r = 0; r < 8; ++r) {
        int row = q0 + t * 16 + r + hlf * 8;
        Out[((size_t)b * NS + row) * ND + col] = acc[t][c][r] / lr[r];
      }
    }
  }
}

// ===================================================================
extern "C" void kernel_launch(void* const* d_in, const int* in_sizes, int n_in,
                              void* d_out, int out_size, void* d_ws, size_t ws_size,
                              hipStream_t stream) {
  (void)in_sizes; (void)n_in; (void)out_size; (void)ws_size;
  const float* enc = (const float*)d_in[0];
  const float* Wq  = (const float*)d_in[1];
  const float* Wk  = (const float*)d_in[2];
  float* out = (float*)d_out;

  // workspace: Qb (2MB) | Kb (2MB) | Vb (32MB), all bf16
  bf16_t* Qb = (bf16_t*)d_ws;
  bf16_t* Kb = Qb + (size_t)NB * NS * NDQK;
  bf16_t* Vb = Kb + (size_t)NB * NS * NDQK;

  const size_t totalV = (size_t)NB * NS * ND;         // 16.7M elems
  convert_v_kernel<<<(unsigned)(totalV / (256 * 8)), 256, 0, stream>>>(enc, Vb);
  proj_kernel<<<dim3(NB * NS / 16, 2), 128, 0, stream>>>(enc, Wq, Wk, Qb, Kb);
  flash_kernel<<<dim3(NS / BQ, NB, ND / DN), 256, 0, stream>>>(Qb, Kb, Vb, out);
}
